// ProgressiveCrossAttention_11192684773748
// MI455X (gfx1250) — compile-verified, matrix operand order checked by reading, not run on hardware
//
#include <hip/hip_runtime.h>

// ---------- types ----------
typedef __attribute__((ext_vector_type(16))) __bf16         v16bf;
typedef __attribute__((ext_vector_type(8)))  float          v8f;
typedef __attribute__((ext_vector_type(8)))  unsigned short u16x8;
typedef __attribute__((ext_vector_type(4)))  unsigned int   u32x4;
typedef __attribute__((ext_vector_type(8)))  int            i32x8;
typedef __attribute__((ext_vector_type(4)))  int            i32x4;

union FragAB { v16bf v; u16x8 h[2]; };

#define BM 128
#define BN 128
#define BK 32

// f32 -> bf16, round-to-nearest-even
__device__ __forceinline__ unsigned short f2bf(float f) {
    union { float f; unsigned u; } x; x.f = f;
    unsigned r = 0x7FFFu + ((x.u >> 16) & 1u);
    return (unsigned short)((x.u + r) >> 16);
}

// LDS byte address of a __shared__ object (low 32 bits of the flat address)
__device__ __forceinline__ unsigned lds_addr_of(const void* p) {
    return (unsigned)(unsigned long long)p;
}

// ---------- Tensor Data Mover: 2D tile (tile0 x tile1 elems, 2B/elem) -> LDS ----------
// D# per CDNA5 ISA: group0 = {flags, lds_addr, global_addr, type=2},
// group1 = {data_size=2B, tensor_dim0/1, tile_dim0/1, tensor_dim0_stride}.
__device__ __forceinline__ void tdm_load_tile_bf16(unsigned lds_addr,
                                                   unsigned long long gaddr,
                                                   unsigned td0, unsigned td1,
                                                   unsigned tile0, unsigned tile1,
                                                   unsigned stride0) {
    u32x4 g0;
    g0[0] = 1u;                                      // count=1, user descriptor
    g0[1] = lds_addr;                                // lds_addr (bytes)
    g0[2] = (unsigned)(gaddr & 0xFFFFFFFFu);         // global_addr[31:0]
    g0[3] = (unsigned)((gaddr >> 32) & 0x01FFFFFFu)  // global_addr[56:32]
            | 0x80000000u;                           // type=2 ("image")

    i32x8 g1;
    g1[0] = (int)0x00010000u;                        // data_size=1 (2 bytes), no mcast/pad
    g1[1] = (int)((td0 & 0xFFFFu) << 16);            // tensor_dim0[15:0]
    g1[2] = (int)((td0 >> 16) | ((td1 & 0xFFFFu) << 16)); // td0[31:16] | td1[15:0]
    g1[3] = (int)((td1 >> 16) | (tile0 << 16));      // td1[31:16] | tile_dim0
    g1[4] = (int)(tile1 & 0xFFFFu);                  // tile_dim1 (tile_dim2=0)
    g1[5] = (int)stride0;                            // tensor_dim0_stride[31:0]
    g1[6] = 0;
    g1[7] = 0;

    i32x4 gz = {};
#if defined(__clang_major__) && (__clang_major__ >= 23)
    i32x8 gz8 = {};
    __builtin_amdgcn_tensor_load_to_lds(g0, g1, gz, gz, gz8, 0);
#else
    __builtin_amdgcn_tensor_load_to_lds(g0, g1, gz, gz, 0);
#endif
}

// ---------- W (f32, row-major KxN) -> Wt (bf16, NxK) ----------
__global__ void transpose_f32_to_bf16(const float* __restrict__ in,
                                      unsigned short* __restrict__ out,
                                      int rows, int cols) {
    __shared__ float t[32][33];
    int tx = threadIdx.x, ty = threadIdx.y;
    int r0 = blockIdx.y * 32, c0 = blockIdx.x * 32;
    for (int i = ty; i < 32; i += 8)
        t[i][tx] = in[(size_t)(r0 + i) * cols + c0 + tx];
    __syncthreads();
    for (int i = ty; i < 32; i += 8)
        out[(size_t)(c0 + i) * rows + r0 + tx] = f2bf(t[tx][i]);
}

// ---------- V (bf16, rows x cols) -> Vt (bf16, cols x rows), batched ----------
__global__ void transpose_bf16_k(const unsigned short* __restrict__ in,
                                 unsigned short* __restrict__ out,
                                 int rows, int cols, long long sIn, long long sOut) {
    __shared__ unsigned short t[32][33];
    int z = blockIdx.z;
    in  += (size_t)z * sIn;
    out += (size_t)z * sOut;
    int tx = threadIdx.x, ty = threadIdx.y;
    int r0 = blockIdx.y * 32, c0 = blockIdx.x * 32;
    for (int i = ty; i < 32; i += 8)
        t[i][tx] = in[(size_t)(r0 + i) * cols + c0 + tx];
    __syncthreads();
    for (int i = ty; i < 32; i += 8)
        out[(size_t)(c0 + i) * rows + r0 + tx] = t[tx][i];
}

// ---------- generic WMMA GEMM: C = A[MxK] * Bt[NxK]^T (+ epilogue) ----------
// MODE 0: store bf16          (projections -> Q,K,V)
// MODE 1: store f32 * alpha   (scores)
// MODE 2: store f32 + Add     (attn @ V + left)
// A_IS_F32: A read as f32, converted to bf16 in VALU during LDS staging (sync path).
// !A_IS_F32: both operands bf16 -> double-buffered TDM tensor_load_to_lds pipeline.
template<int MODE, bool A_IS_F32>
__global__ __launch_bounds__(256)
void gemm_bt_wmma(const void* __restrict__ A_, const unsigned short* __restrict__ Bt,
                  void* __restrict__ C_, const float* __restrict__ Add,
                  int N, int K, int Mrows,
                  long long sA, long long sB, long long sC, long long sAdd,
                  float alpha) {
    __shared__ __align__(16) unsigned short ldsA[2][BM * BK];
    __shared__ __align__(16) unsigned short ldsB[2][BN * BK];

    const int tid  = threadIdx.x;
    const int wid  = tid >> 5;          // 8 wave32s
    const int lane = tid & 31;
    const int r    = lane & 15;
    const int hf   = lane >> 4;
    const int z    = blockIdx.z;
    const int m0   = blockIdx.y * BM;
    const int n0   = blockIdx.x * BN;

    const unsigned short* Bz = Bt + (size_t)z * sB;

    v8f acc[8] = {};

    if constexpr (A_IS_F32) {
        // ---------------- sync path: f32 A converted during staging ----------------
        const float* Az = (const float*)A_ + (size_t)z * sA;
        for (int kk = 0; kk < K; kk += BK) {
            __syncthreads();
            #pragma unroll
            for (int it = 0; it < (BM * BK / 8) / 256; ++it) {
                int s = tid + it * 256;
                int row = s >> 2, c = (s & 3) << 3;
                const float4* s4 = (const float4*)(Az + (size_t)(m0 + row) * K + kk + c);
                float4 f0 = s4[0], f1 = s4[1];
                u16x8 o;
                o[0] = f2bf(f0.x); o[1] = f2bf(f0.y); o[2] = f2bf(f0.z); o[3] = f2bf(f0.w);
                o[4] = f2bf(f1.x); o[5] = f2bf(f1.y); o[6] = f2bf(f1.z); o[7] = f2bf(f1.w);
                *(u16x8*)&ldsA[0][row * BK + c] = o;
            }
            #pragma unroll
            for (int it = 0; it < (BN * BK / 8) / 256; ++it) {
                int s = tid + it * 256;
                int row = s >> 2, c = (s & 3) << 3;
                *(u16x8*)&ldsB[0][row * BK + c] =
                    *(const u16x8*)(Bz + (size_t)(n0 + row) * K + kk + c);
            }
            __syncthreads();

            FragAB a;
            a.h[0] = *(const u16x8*)&ldsA[0][(wid * 16 + r) * BK + hf * 8];
            a.h[1] = *(const u16x8*)&ldsA[0][(wid * 16 + r) * BK + 16 + hf * 8];
            FragAB b[8];
            #pragma unroll
            for (int nt = 0; nt < 8; ++nt) {
                b[nt].h[0] = *(const u16x8*)&ldsB[0][(nt * 16 + r) * BK + hf * 8];
                b[nt].h[1] = *(const u16x8*)&ldsB[0][(nt * 16 + r) * BK + 16 + hf * 8];
            }
            #pragma unroll
            for (int nt = 0; nt < 8; ++nt)
                acc[nt] = __builtin_amdgcn_wmma_f32_16x16x32_bf16(
                    false, a.v, false, b[nt].v, (short)0, acc[nt], false, false);
        }
    } else {
        // ---------------- TDM path: double-buffered tensor_load_to_lds ----------------
        const unsigned short* Az = (const unsigned short*)A_ + (size_t)z * sA;
        const unsigned long long gA = (unsigned long long)(Az + (size_t)m0 * K);
        const unsigned long long gB = (unsigned long long)(Bz + (size_t)n0 * K);
        const unsigned aAddr0 = lds_addr_of(&ldsA[0][0]);
        const unsigned aAddr1 = lds_addr_of(&ldsA[1][0]);
        const unsigned bAddr0 = lds_addr_of(&ldsB[0][0]);
        const unsigned bAddr1 = lds_addr_of(&ldsB[1][0]);
        const int nIter = K / BK;

        if (wid == 0) {
            tdm_load_tile_bf16(aAddr0, gA, (unsigned)K, (unsigned)Mrows, BK, BM, (unsigned)K);
            tdm_load_tile_bf16(bAddr0, gB, (unsigned)K, (unsigned)N,     BK, BN, (unsigned)K);
        }

        for (int i = 0; i < nIter; ++i) {
            const int cur = i & 1;
            if (wid == 0) {
                if (i + 1 < nIter) {
                    unsigned long long off = (unsigned long long)(i + 1) * (BK * 2);
                    tdm_load_tile_bf16(cur ? aAddr0 : aAddr1, gA + off,
                                       (unsigned)K, (unsigned)Mrows, BK, BM, (unsigned)K);
                    tdm_load_tile_bf16(cur ? bAddr0 : bAddr1, gB + off,
                                       (unsigned)K, (unsigned)N,     BK, BN, (unsigned)K);
                    __builtin_amdgcn_s_wait_tensorcnt(2);  // tiles for iter i have landed
                } else {
                    __builtin_amdgcn_s_wait_tensorcnt(0);
                }
            }
            __syncthreads();   // buf[cur] visible to all waves

            const unsigned short* lA = &ldsA[cur][0];
            const unsigned short* lB = &ldsB[cur][0];
            FragAB a;
            a.h[0] = *(const u16x8*)&lA[(wid * 16 + r) * BK + hf * 8];
            a.h[1] = *(const u16x8*)&lA[(wid * 16 + r) * BK + 16 + hf * 8];
            FragAB b[8];
            #pragma unroll
            for (int nt = 0; nt < 8; ++nt) {
                b[nt].h[0] = *(const u16x8*)&lB[(nt * 16 + r) * BK + hf * 8];
                b[nt].h[1] = *(const u16x8*)&lB[(nt * 16 + r) * BK + 16 + hf * 8];
            }
            #pragma unroll
            for (int nt = 0; nt < 8; ++nt)
                acc[nt] = __builtin_amdgcn_wmma_f32_16x16x32_bf16(
                    false, a.v, false, b[nt].v, (short)0, acc[nt], false, false);

            __syncthreads();   // all reads of buf[cur] done before it is refilled
        }
    }

    // ---- epilogue: C/D layout: VGPR e -> M = e + 8*hf, lane r -> N ----
    #pragma unroll
    for (int nt = 0; nt < 8; ++nt) {
        #pragma unroll
        for (int e = 0; e < 8; ++e) {
            int row = m0 + wid * 16 + e + hf * 8;
            int col = n0 + nt * 16 + r;
            float v = acc[nt][e];
            if constexpr (MODE == 0) {
                unsigned short* C = (unsigned short*)C_ + (size_t)z * sC;
                C[(size_t)row * N + col] = f2bf(v);
            } else if constexpr (MODE == 1) {
                float* C = (float*)C_ + (size_t)z * sC;
                C[(size_t)row * N + col] = v * alpha;
            } else {
                float* C = (float*)C_ + (size_t)z * sC;
                const float* Ad = Add + (size_t)z * sAdd;
                C[(size_t)row * N + col] = v + Ad[(size_t)row * N + col];
            }
        }
    }
}

// ---------- row softmax: S f32 [rows x 4096] -> P bf16 ----------
__global__ __launch_bounds__(256)
void softmax_rows(const float* __restrict__ S, unsigned short* __restrict__ P, int ncols) {
    __shared__ float red[256];
    const int row = blockIdx.x;
    const int tid = threadIdx.x;
    const float* src = S + (size_t)row * ncols;

    float v[16];
    float m = -3.4e38f;
    #pragma unroll
    for (int i = 0; i < 16; ++i) { v[i] = src[tid + i * 256]; m = fmaxf(m, v[i]); }
    red[tid] = m; __syncthreads();
    for (int off = 128; off > 0; off >>= 1) {
        if (tid < off) red[tid] = fmaxf(red[tid], red[tid + off]);
        __syncthreads();
    }
    m = red[0]; __syncthreads();

    float s = 0.f;
    #pragma unroll
    for (int i = 0; i < 16; ++i) { v[i] = __expf(v[i] - m); s += v[i]; }
    red[tid] = s; __syncthreads();
    for (int off = 128; off > 0; off >>= 1) {
        if (tid < off) red[tid] += red[tid + off];
        __syncthreads();
    }
    float inv = 1.0f / red[0];

    unsigned short* dst = P + (size_t)row * ncols;
    #pragma unroll
    for (int i = 0; i < 16; ++i) dst[tid + i * 256] = f2bf(v[i] * inv);
}

// ---------- launch ----------
extern "C" void kernel_launch(void* const* d_in, const int* in_sizes, int n_in,
                              void* d_out, int out_size, void* d_ws, size_t ws_size,
                              hipStream_t stream) {
    const float* left  = (const float*)d_in[0];
    const float* right = (const float*)d_in[1];
    const float* Wq    = (const float*)d_in[2];
    const float* Wk    = (const float*)d_in[3];
    const float* Wv    = (const float*)d_in[4];
    float* out = (float*)d_out;

    const int B = 4, S = 4096, D = 1024;
    const long long SD = (long long)S * D;       // per-batch q/k/v elems
    const long long SS = (long long)S * S;       // per-batch score elems
    const long long BSD = (long long)B * SD;

    char* ws = (char*)d_ws;
    size_t off = 0;
    auto alloc = [&](size_t bytes) -> void* {
        void* p = ws + off;
        off += (bytes + 255) & ~(size_t)255;
        return p;
    };
    unsigned short* Wqt = (unsigned short*)alloc((size_t)D * D * 2);
    unsigned short* Wkt = (unsigned short*)alloc((size_t)D * D * 2);
    unsigned short* Wvt = (unsigned short*)alloc((size_t)D * D * 2);
    unsigned short* Q   = (unsigned short*)alloc((size_t)BSD * 2);
    unsigned short* Kb  = (unsigned short*)alloc((size_t)BSD * 2);
    unsigned short* V   = (unsigned short*)alloc((size_t)BSD * 2);
    unsigned short* Vt  = (unsigned short*)alloc((size_t)BSD * 2);
    float*          Sm  = (float*)alloc((size_t)B * SS * 4);
    unsigned short* P   = (unsigned short*)alloc((size_t)B * SS * 2);

    dim3 tb(32, 8), blk(256);

    // 1) weight transpose + bf16 convert
    transpose_f32_to_bf16<<<dim3(D / 32, D / 32), tb, 0, stream>>>(Wq, Wqt, D, D);
    transpose_f32_to_bf16<<<dim3(D / 32, D / 32), tb, 0, stream>>>(Wk, Wkt, D, D);
    transpose_f32_to_bf16<<<dim3(D / 32, D / 32), tb, 0, stream>>>(Wv, Wvt, D, D);

    // 2) projections: Q=left@Wq, K=right@Wk, V=right@Wv   (bf16 out, f32-A sync path)
    gemm_bt_wmma<0, true><<<dim3(D / BN, (B * S) / BM, 1), blk, 0, stream>>>(
        left, Wqt, Q, nullptr, D, D, B * S, 0, 0, 0, 0, 1.0f);
    gemm_bt_wmma<0, true><<<dim3(D / BN, (B * S) / BM, 1), blk, 0, stream>>>(
        right, Wkt, Kb, nullptr, D, D, B * S, 0, 0, 0, 0, 1.0f);
    gemm_bt_wmma<0, true><<<dim3(D / BN, (B * S) / BM, 1), blk, 0, stream>>>(
        right, Wvt, V, nullptr, D, D, B * S, 0, 0, 0, 0, 1.0f);

    // 3) V -> Vt (per batch), so attn@V uses the uniform Bt[N][K] path
    transpose_bf16_k<<<dim3(D / 32, S / 32, B), tb, 0, stream>>>(V, Vt, S, D, SD, SD);

    // 4) scores: Sm = (Q @ K^T) * D^-0.5   (f32 out, batched, TDM pipeline)
    gemm_bt_wmma<1, false><<<dim3(S / BN, S / BM, B), blk, 0, stream>>>(
        Q, Kb, Sm, nullptr, S, D, S, SD, SD, SS, 0, 0.03125f);

    // 5) row softmax -> P (bf16)
    softmax_rows<<<dim3(B * S), blk, 0, stream>>>(Sm, P, S);

    // 6) out = P @ V + left   (f32 out, batched, TDM pipeline)
    gemm_bt_wmma<2, false><<<dim3(D / BN, S / BM, B), blk, 0, stream>>>(
        P, Vt, out, left, D, S, S, SS, SD, SD, SD, 1.0f);
}
